// MultiScaleRetention_18021682774290
// MI455X (gfx1250) — compile-verified, matrix operand order checked by reading, not run on hardware
//
#include <hip/hip_runtime.h>
#include <hip/hip_bf16.h>
#include <math.h>

// ---------------------------------------------------------------------------
// MultiScaleRetention for MI455X (gfx1250): bf16 WMMA pipeline, f32 accum,
// async-LDS (ASYNCcnt) staged GEMMs.
// B=2, S=2048, H=2048, NH=16, D=128
// ---------------------------------------------------------------------------

#define B_   2
#define S_   2048
#define H_   2048
#define NH_  16
#define D_   128
#define MROWS (B_ * S_)                 // 4096 token rows
#define SCALE_ 0.08838834764831845f    // D^-0.5
#define GN_EPS_ 1e-5f

typedef __bf16 bf16_t;
typedef __attribute__((ext_vector_type(16))) __bf16 v16bf;
typedef __attribute__((ext_vector_type(8)))  __bf16 v8bf;
typedef __attribute__((ext_vector_type(8)))  float  v8f;

// ---------------------------------------------------------------------------
// Fragment helpers (CDNA5 wave32 WMMA VGPR layouts, CDNA5 ISA)
//
// 16-bit A (16x32, MxK) per lane:
//   lanes 0-15 : row M = lane,    K = {k0+0..7,  k0+16..23}
//   lanes 16-31: row M = lane-16, K = {k0+8..15, k0+24..31}
// B (32x16, KxN) mirrors with lane -> column N; same K distribution.
// ---------------------------------------------------------------------------
static __device__ __forceinline__ v16bf ld_frag16x32(const bf16_t* __restrict__ base,
                                                     int ld, int r0, int k0) {
  const int l   = threadIdx.x & 31;
  const int row = r0 + (l & 15);
  const int kb  = k0 + ((l & 16) >> 1);        // +8 for upper half-wave
  const bf16_t* p = base + (long)row * ld + kb;
  v8bf lo = *(const v8bf*)(p);                 // K = kb .. kb+7
  v8bf hi = *(const v8bf*)(p + 16);            // K = kb+16 .. kb+23
  v16bf r;
#pragma unroll
  for (int i = 0; i < 8; ++i) { r[i] = lo[i]; r[8 + i] = hi[i]; }
  return r;
}

static __device__ __forceinline__ v8f wmma_bf16(v16bf a, v16bf b, v8f c) {
  return __builtin_amdgcn_wmma_f32_16x16x32_bf16(false, a, false, b, (short)0, c,
                                                 false, false);
}

// C/D f32 16x16 layout: VGPR j -> M = j (lanes 0-15) / 8+j (lanes 16-31); N = lane&15
static __device__ __forceinline__ void st_tile(float* __restrict__ Cf,
                                               bf16_t* __restrict__ Cb,
                                               long ldc, int m0, int n0, v8f acc) {
  const int l   = threadIdx.x & 31;
  const int col = n0 + (l & 15);
  const int r0  = m0 + ((l & 16) >> 1);
#pragma unroll
  for (int j = 0; j < 8; ++j) {
    if (Cf) Cf[(long)(r0 + j) * ldc + col] = acc[j];
    else    Cb[(long)(r0 + j) * ldc + col] = (bf16_t)acc[j];
  }
}

// ---------------------------------------------------------------------------
// Async global -> LDS copy (CDNA5 ISA; ASYNCcnt tracked).
// LDS aperture maps addr[31:0] -> LDS address (ISA 10.2), so the low 32 bits
// of a generic pointer to a __shared__ object are the LDS byte address.
// ---------------------------------------------------------------------------
static __device__ __forceinline__ void async_b128(unsigned lds_addr,
                                                  const bf16_t* g) {
  asm volatile("global_load_async_to_lds_b128 %0, %1, off"
               :: "v"(lds_addr), "v"(g)
               : "memory");
}
static __device__ __forceinline__ void wait_async0() {
  asm volatile("s_wait_asynccnt 0x0" ::: "memory");
}

// ---------------------------------------------------------------------------
// Init: per-head log2(gamma_h), gamma = 1 - exp(linspace(log 1/32, log 1/512))
// ---------------------------------------------------------------------------
__global__ void init_gamma_kernel(float* __restrict__ lg) {
  int h = threadIdx.x;
  if (h < NH_) {
    float a = logf(1.0f / 32.0f), b = logf(1.0f / 512.0f);
    float t = a + (b - a) * ((float)h / 15.0f);
    lg[h] = log2f(1.0f - expf(t));
  }
}

__global__ void cast_f32_bf16_kernel(const float* __restrict__ src,
                                     bf16_t* __restrict__ dst, long n) {
  long i = (long)blockIdx.x * blockDim.x + threadIdx.x;
  if (i < n) dst[i] = (bf16_t)src[i];
}

// vb[(b*S+m)*H + h*D + d] -> vbT[((b*NH+h)*D + d)*S + m]   (contiguous keys)
__global__ void transpose_v_kernel(const bf16_t* __restrict__ vb,
                                   bf16_t* __restrict__ vbT) {
  long i = (long)blockIdx.x * blockDim.x + threadIdx.x;
  if (i >= (long)MROWS * H_) return;
  int  d  = (int)(i & 127);
  int  h  = (int)((i >> 7) & 15);
  long sm = i >> 11;
  int  m  = (int)(sm & (S_ - 1));
  int  b  = (int)(sm >> 11);
  vbT[(((long)(b * NH_ + h)) * D_ + d) * S_ + m] = vb[i];
}

// ---------------------------------------------------------------------------
// GEMM: C[4096 x N] = A[4096 x K](bf16) @ W[N x K]^T (bf16), f32 accum.
// Block tile 128x64; K slab 64, double-buffered async-LDS staging:
//   sA 128x64 bf16 (16KB), sB 64x64 bf16 (8KB), x2 buffers = 48KB LDS.
// 8 waves, each wave 32x32 output (2x2 WMMA tiles), 8 WMMAs per K slab.
// ---------------------------------------------------------------------------
__global__ __launch_bounds__(256) void gemm_xwT_kernel(
    const bf16_t* __restrict__ A, const bf16_t* __restrict__ W,
    float* __restrict__ Cf, bf16_t* __restrict__ Cb, int Kdim, int Ndim) {
  __shared__ bf16_t sA[2][128][64];
  __shared__ bf16_t sB[2][64][64];

  const int n0  = blockIdx.x * 64;
  const int m0  = blockIdx.y * 128;
  const int tid = threadIdx.x;
  const int w   = tid >> 5;
  const int wr  = w & 3, wc = w >> 2;
  const int mrow = m0 + wr * 32;
  const int ncol = n0 + wc * 32;

  // Stage one K slab (64 deep) into LDS buffer `buf` with async B128 copies.
  auto stage = [&](int buf, int k0) {
    const unsigned ldsA = (unsigned)(uintptr_t)&sA[buf][0][0];
    const unsigned ldsB = (unsigned)(uintptr_t)&sB[buf][0][0];
#pragma unroll
    for (int j = 0; j < 4; ++j) {                 // A: 1024 x 16B chunks
      const int c = tid + 256 * j;
      const int row = c >> 3, col = (c & 7) * 8;  // col in elements
      async_b128(ldsA + (unsigned)(row * 128 + col * 2),
                 A + (long)(m0 + row) * Kdim + k0 + col);
    }
#pragma unroll
    for (int j = 0; j < 2; ++j) {                 // B: 512 x 16B chunks
      const int c = tid + 256 * j;
      const int row = c >> 3, col = (c & 7) * 8;
      async_b128(ldsB + (unsigned)(row * 128 + col * 2),
                 W + (long)(n0 + row) * Kdim + k0 + col);
    }
  };

  v8f acc00 = {}, acc01 = {}, acc10 = {}, acc11 = {};

  stage(0, 0);
  wait_async0();
  __syncthreads();

  for (int k0 = 0; k0 < Kdim; k0 += 64) {
    const int buf = (k0 >> 6) & 1;
    if (k0 + 64 < Kdim) stage(buf ^ 1, k0 + 64);  // prefetch next slab

    const bf16_t* tA = &sA[buf][0][0];
    const bf16_t* tB = &sB[buf][0][0];
#pragma unroll
    for (int ks = 0; ks < 64; ks += 32) {
      v16bf a0 = ld_frag16x32(tA, 64, wr * 32,      ks);
      v16bf a1 = ld_frag16x32(tA, 64, wr * 32 + 16, ks);
      v16bf b0 = ld_frag16x32(tB, 64, wc * 32,      ks);
      v16bf b1 = ld_frag16x32(tB, 64, wc * 32 + 16, ks);
      acc00 = wmma_bf16(a0, b0, acc00);
      acc01 = wmma_bf16(a0, b1, acc01);
      acc10 = wmma_bf16(a1, b0, acc10);
      acc11 = wmma_bf16(a1, b1, acc11);
    }
    wait_async0();      // own async copies into next buffer complete
    __syncthreads();    // everyone done reading cur + done filling next
  }

  st_tile(Cf, Cb, Ndim, mrow,      ncol,      acc00);
  st_tile(Cf, Cb, Ndim, mrow,      ncol + 16, acc01);
  st_tile(Cf, Cb, Ndim, mrow + 16, ncol,      acc10);
  st_tile(Cf, Cb, Ndim, mrow + 16, ncol + 16, acc11);
}

// ---------------------------------------------------------------------------
// Retention: per (b,h), 64 query rows per block. 8 waves: wr=rows(16),
// wc=output d-half(64). Causal m-block loop; scores via 4 chained WMMAs,
// decay applied per accumulator element, LDS round-trip to A-layout,
// then scores @ V via 2 WMMAs per d-tile.
// ---------------------------------------------------------------------------
__global__ __launch_bounds__(256) void retention_kernel(
    const bf16_t* __restrict__ qb, const bf16_t* __restrict__ kb,
    const bf16_t* __restrict__ vbT, const float* __restrict__ lg,
    float* __restrict__ O) {
  __shared__ bf16_t sS[8][16][64];   // wave-private score tiles (bf16)

  const int nblk = blockIdx.x;      // 0..31 (64-row blocks)
  const int h    = blockIdx.y;
  const int b    = blockIdx.z;
  const int w    = threadIdx.x >> 5;
  const int wr   = w & 3, wc = w >> 2;
  const int nb   = nblk * 64 + wr * 16;

  const bf16_t* qbase = qb  + (long)b * S_ * H_ + (long)h * D_;   // [n][d], ld=H
  const bf16_t* kbase = kb  + (long)b * S_ * H_ + (long)h * D_;   // [m][d], ld=H
  const bf16_t* vtb   = vbT + (long)(b * NH_ + h) * D_ * S_;      // [d][m], ld=S
  const float   l2g   = lg[h];

  v16bf aQ[4];
#pragma unroll
  for (int kk = 0; kk < 4; ++kk) aQ[kk] = ld_frag16x32(qbase, H_, nb, kk * 32);

  v8f accO[4] = {v8f{}, v8f{}, v8f{}, v8f{}};

  const int l    = threadIdx.x & 31;
  const int lcol = l & 15;
  const int lrb  = (l & 16) >> 1;

  for (int mb = 0; mb <= nblk * 64; mb += 64) {
    // ---- scores: Q[16x128] x K^T -> 4 sub-tiles of 16 keys each ----
#pragma unroll
    for (int st = 0; st < 4; ++st) {
      v8f s = {};
#pragma unroll
      for (int kk = 0; kk < 4; ++kk)
        s = wmma_bf16(aQ[kk], ld_frag16x32(kbase, H_, mb + st * 16, kk * 32), s);
      const int mcol = mb + st * 16 + lcol;   // global key index of this lane
      const int nr0  = nb + lrb;              // global query row base
#pragma unroll
      for (int j = 0; j < 8; ++j) {
        const int n = nr0 + j;
        const float wgt = (n >= mcol)
            ? SCALE_ * exp2f(l2g * (float)(n - mcol)) : 0.0f;
        sS[w][lrb + j][st * 16 + lcol] = (bf16_t)(s[j] * wgt);
      }
    }
    // wave-private LDS region: same-wave RAW, hardware DScnt ordering applies

    // ---- O += scores[16x64] @ V[64 x d] ----
    v16bf aS0 = ld_frag16x32(&sS[w][0][0], 64, 0, 0);
    v16bf aS1 = ld_frag16x32(&sS[w][0][0], 64, 0, 32);
#pragma unroll
    for (int t = 0; t < 4; ++t) {
      const int d0 = wc * 64 + t * 16;
      v16bf bv0 = ld_frag16x32(vtb, S_, d0, mb);        // B(k=m, n=d) = V^T[d][m]
      v16bf bv1 = ld_frag16x32(vtb, S_, d0, mb + 32);
      accO[t] = wmma_bf16(aS0, bv0, accO[t]);
      accO[t] = wmma_bf16(aS1, bv1, accO[t]);
    }
  }

  float* obase = O + (long)b * S_ * H_ + (long)h * D_;
#pragma unroll
  for (int t = 0; t < 4; ++t)
    st_tile(obase, (bf16_t*)nullptr, H_, nb, wc * 64 + t * 16, accO[t]);
}

// ---------------------------------------------------------------------------
// GroupNorm over each (b,h) group of D*S elements, fused with silu gating.
// ---------------------------------------------------------------------------
__global__ __launch_bounds__(256) void gn_gate_kernel(
    const float* __restrict__ O, const float* __restrict__ Graw,
    const float* __restrict__ gnw, const float* __restrict__ gnb,
    bf16_t* __restrict__ Gated) {
  const int g = blockIdx.x;
  const int b = g >> 4, h = g & 15;
  const float* base  = O    + (long)b * S_ * H_ + (long)h * D_;  // [n][d], ld=H
  const float* gbase = Graw + (long)b * S_ * H_ + (long)h * D_;
  bf16_t*      obase = Gated + (long)b * S_ * H_ + (long)h * D_;

  float s = 0.f, ss = 0.f;
  for (int idx = threadIdx.x; idx < S_ * D_; idx += 256) {
    const int n = idx >> 7, d = idx & 127;
    const float v = base[(long)n * H_ + d];
    s += v; ss += v * v;
  }
  __shared__ float rs[256], rss[256];
  rs[threadIdx.x] = s; rss[threadIdx.x] = ss;
  __syncthreads();
  for (int stp = 128; stp > 0; stp >>= 1) {
    if (threadIdx.x < stp) {
      rs[threadIdx.x]  += rs[threadIdx.x + stp];
      rss[threadIdx.x] += rss[threadIdx.x + stp];
    }
    __syncthreads();
  }
  const float cnt = (float)(S_ * D_);
  const float mu  = rs[0] / cnt;
  const float var = rss[0] / cnt - mu * mu;
  const float inv = rsqrtf(var + GN_EPS_);

  for (int idx = threadIdx.x; idx < S_ * D_; idx += 256) {
    const int n = idx >> 7, d = idx & 127;
    const int c = h * D_ + d;
    const float y  = (base[(long)n * H_ + d] - mu) * inv * gnw[c] + gnb[c];
    const float x  = gbase[(long)n * H_ + d];
    const float si = x / (1.0f + expf(-x));      // silu
    obase[(long)n * H_ + d] = (bf16_t)(si * y);
  }
}

// ---------------------------------------------------------------------------
// Launch
// ---------------------------------------------------------------------------
extern "C" void kernel_launch(void* const* d_in, const int* in_sizes, int n_in,
                              void* d_out, int out_size, void* d_ws, size_t ws_size,
                              hipStream_t stream) {
  const float* x   = (const float*)d_in[0];
  const float* Ww[5] = {(const float*)d_in[1], (const float*)d_in[2],
                        (const float*)d_in[3], (const float*)d_in[4],
                        (const float*)d_in[5]};
  const float* gnw = (const float*)d_in[6];
  const float* gnb = (const float*)d_in[7];
  float* out = (float*)d_out;

  char* ws = (char*)d_ws;
  size_t off = 0;
  auto alloc = [&](size_t bytes) -> void* {
    void* p = ws + off;
    off += (bytes + 255) & ~(size_t)255;
    return p;
  };
  const long NX = (long)MROWS * H_;       // 8388608 activation elements
  const long NW = (long)H_ * H_;          // 4194304 weight elements

  bf16_t* xb   = (bf16_t*)alloc(NX * 2);
  bf16_t* wb[5];
  for (int i = 0; i < 5; ++i) wb[i] = (bf16_t*)alloc(NW * 2);
  bf16_t* qb   = (bf16_t*)alloc(NX * 2);
  bf16_t* kbuf = (bf16_t*)alloc(NX * 2);
  bf16_t* vb   = (bf16_t*)alloc(NX * 2);
  bf16_t* vbT  = (bf16_t*)alloc(NX * 2);
  float*  gf   = (float*) alloc(NX * 4);
  float*  of   = (float*) alloc(NX * 4);
  bf16_t* gb   = (bf16_t*)alloc(NX * 2);
  float*  lg   = (float*) alloc(NH_ * 4);
  (void)ws_size; (void)in_sizes; (void)n_in; (void)out_size;

  init_gamma_kernel<<<1, 32, 0, stream>>>(lg);

  const int castB = 256;
  cast_f32_bf16_kernel<<<(int)((NX + castB - 1) / castB), castB, 0, stream>>>(x, xb, NX);
  for (int i = 0; i < 5; ++i)
    cast_f32_bf16_kernel<<<(int)((NW + castB - 1) / castB), castB, 0, stream>>>(Ww[i], wb[i], NW);

  dim3 gg(H_ / 64, MROWS / 128);  // 32 x 32 blocks
  gemm_xwT_kernel<<<gg, 256, 0, stream>>>(xb, wb[0], nullptr, qb,   H_, H_); // Q
  gemm_xwT_kernel<<<gg, 256, 0, stream>>>(xb, wb[1], nullptr, kbuf, H_, H_); // K
  gemm_xwT_kernel<<<gg, 256, 0, stream>>>(xb, wb[2], nullptr, vb,   H_, H_); // V
  gemm_xwT_kernel<<<gg, 256, 0, stream>>>(xb, wb[3], gf, nullptr,   H_, H_); // G (pre-silu)

  transpose_v_kernel<<<(int)((NX + castB - 1) / castB), castB, 0, stream>>>(vb, vbT);

  dim3 rg(S_ / 64, NH_, B_);
  retention_kernel<<<rg, 256, 0, stream>>>(qb, kbuf, vbT, lg, of);

  gn_gate_kernel<<<B_ * NH_, 256, 0, stream>>>(of, gf, gnw, gnb, gb);

  gemm_xwT_kernel<<<gg, 256, 0, stream>>>(gb, wb[4], out, nullptr, H_, H_);  // Wo
}
